// CosineSelfAttention_2525440770111
// MI455X (gfx1250) — compile-verified, matrix operand order checked by reading, not run on hardware
//
#include <hip/hip_runtime.h>
#include <math.h>

typedef __bf16 bf16_t;
typedef bf16_t v8bf  __attribute__((ext_vector_type(8)));
typedef bf16_t v16bf __attribute__((ext_vector_type(16)));
typedef float  v8f   __attribute__((ext_vector_type(8)));
typedef float  v2f   __attribute__((ext_vector_type(2)));

constexpr int B_ = 4, S_ = 4096, DM_ = 1024, H_ = 16, D_ = 64;
constexpr float EPS_ = 1e-5f;

// ---------------- utility kernels ----------------
__global__ void k_zero(float* p, int n) {
  int i = blockIdx.x * 256 + threadIdx.x;
  if (i < n) p[i] = 0.0f;
}

__global__ void k_cvt(const float* __restrict__ in, bf16_t* __restrict__ out, int n) {
  int i = blockIdx.x * 256 + threadIdx.x;
  if (i < n) out[i] = (bf16_t)in[i];
}

// W is [K=1024][N=1024] row-major; produce Wt[N][K] in bf16 so B-matrix tiles
// load with the same per-lane K-slice pattern as A-matrix tiles.
__global__ void k_cvt_t(const float* __restrict__ in, bf16_t* __restrict__ out) {
  int i = blockIdx.x * 256 + threadIdx.x;      // 1M elements
  int n = i >> 10, k = i & 1023;
  out[((size_t)n << 10) | k] = (bf16_t)in[((size_t)k << 10) | n];
}

// Load one 16x32 bf16 operand tile per ISA 16-bit A layout:
// lane<16 -> K = kb0..kb0+7 and +16..+23 ; lane>=16 -> +8.. and +24..
__device__ __forceinline__ v16bf ld_bf16_tile(const bf16_t* rowp, int kbase) {
  v8bf lo = *(const v8bf*)(rowp + kbase);
  v8bf hi = *(const v8bf*)(rowp + kbase + 16);
  return __builtin_shufflevector(lo, hi, 0,1,2,3,4,5,6,7,8,9,10,11,12,13,14,15);
}

// ---------------- fused QKV projection GEMM (bf16 WMMA, f32 accum) ----------------
// grid: 1536 blocks x 256 threads = 12288 waves; wave = one 64x64 C tile.
// flat = proj(3) x ntile(16) x mtile(256)
__global__ void __launch_bounds__(256) k_gemm_qkv(
    const bf16_t* __restrict__ X, const bf16_t* __restrict__ Wt,
    const float* __restrict__ bq, const float* __restrict__ bk, const float* __restrict__ bv,
    float* __restrict__ Q, float* __restrict__ K, float* __restrict__ V)
{
  const int lane = threadIdx.x & 31;
  const int l15  = lane & 15;
  const int lh   = lane >> 4;
  const int flat = blockIdx.x * 8 + (threadIdx.x >> 5);
  const int mt   = flat & 255;
  const int t    = flat >> 8;
  const int nt   = t & 15;
  const int proj = t >> 4;

  const bf16_t* W    = Wt + (size_t)proj * (DM_ * DM_);
  const float*  bias = (proj == 0) ? bq : (proj == 1) ? bk : bv;
  float*        Out  = (proj == 0) ? Q  : (proj == 1) ? K  : V;

  const int m0 = mt << 6, n0 = nt << 6;
  const bf16_t* Arow[4];
  const bf16_t* Brow[4];
#pragma unroll
  for (int i = 0; i < 4; ++i) Arow[i] = X + (size_t)(m0 + i * 16 + l15) * DM_;
#pragma unroll
  for (int j = 0; j < 4; ++j) Brow[j] = W + (size_t)(n0 + j * 16 + l15) * DM_;

  v8f acc[4][4] = {};
  const int kb0 = lh << 3;   // 0 or 8
  for (int k0 = 0; k0 < DM_; k0 += 32) {
    v16bf a[4], b[4];
#pragma unroll
    for (int i = 0; i < 4; ++i) a[i] = ld_bf16_tile(Arow[i], k0 + kb0);
#pragma unroll
    for (int j = 0; j < 4; ++j) b[j] = ld_bf16_tile(Brow[j], k0 + kb0);
#pragma unroll
    for (int i = 0; i < 4; ++i)
#pragma unroll
      for (int j = 0; j < 4; ++j)
        acc[i][j] = __builtin_amdgcn_wmma_f32_16x16x32_bf16(
            false, a[i], false, b[j], (short)0, acc[i][j], false, false);
  }

  // epilogue: bias add + store. C layout: VGPR r -> M = r + 8*lh, N = l15.
#pragma unroll
  for (int j = 0; j < 4; ++j) {
    const int n = n0 + j * 16 + l15;
    const float bb = bias[n];
#pragma unroll
    for (int i = 0; i < 4; ++i) {
      float* colp = Out + (size_t)(m0 + i * 16 + 8 * lh) * DM_ + n;
#pragma unroll
      for (int r = 0; r < 8; ++r) colp[(size_t)r * DM_] = acc[i][j][r] + bb;
    }
  }
}

// ---------------- mask-scale + cosine normalize Q,K; accumulate ksum/vsum/msum ----------
// one wave per (b,s,h) row of 64; lane handles 2 elements.
__global__ void __launch_bounds__(256) k_normalize(
    float* __restrict__ Q, float* __restrict__ K, const float* __restrict__ V,
    const float* __restrict__ am,
    float* __restrict__ ksum, float* __restrict__ vsum, float* __restrict__ msum)
{
  const int lane = threadIdx.x & 31;
  const int warp = (blockIdx.x * 256 + threadIdx.x) >> 5;   // < B*S*H
  const int h  = warp & (H_ - 1);
  const int bs = warp >> 4;          // b*S + s
  const int b  = bs >> 12;           // S = 4096

  const float m = am[bs] * 1e-4f + 1.0f;
  const size_t off = (size_t)bs * DM_ + h * D_ + lane * 2;

  float2 q = *(const float2*)(Q + off);
  float2 k = *(const float2*)(K + off);
  q.x *= m; q.y *= m; k.x *= m; k.y *= m;
  float ssq = q.x * q.x + q.y * q.y;
  float ssk = k.x * k.x + k.y * k.y;
#pragma unroll
  for (int o = 16; o > 0; o >>= 1) {
    ssq += __shfl_xor(ssq, o, 32);
    ssk += __shfl_xor(ssk, o, 32);
  }
  const float iq = 1.0f / (sqrtf(ssq) + EPS_);
  const float ik = 1.0f / (sqrtf(ssk) + EPS_);
  q.x *= iq; q.y *= iq; k.x *= ik; k.y *= ik;
  *(float2*)(Q + off) = q;
  *(float2*)(K + off) = k;

  const float2 v = *(const float2*)(V + off);
  const int bh = (b << 4) | h;
  atomicAdd(&ksum[bh * D_ + lane * 2    ], k.x);
  atomicAdd(&ksum[bh * D_ + lane * 2 + 1], k.y);
  atomicAdd(&vsum[bh * D_ + lane * 2    ], v.x);
  atomicAdd(&vsum[bh * D_ + lane * 2 + 1], v.y);
  if (h == 0 && lane == 0) atomicAdd(&msum[b], m);
}

// ---------------- KV = K^T V per (b,h), f32 WMMA, S split into 32 chunks of 128 --------
// wave = (bh, chunk); computes full 64x64 partial, atomically accumulates.
__global__ void __launch_bounds__(256) k_kv_accum(
    const float* __restrict__ K, const float* __restrict__ V, float* __restrict__ kv)
{
  const int lane = threadIdx.x & 31;
  const int l15 = lane & 15, lh = lane >> 4;
  const int wid = blockIdx.x * 8 + (threadIdx.x >> 5);   // < 64*32
  const int chunk = wid & 31;
  const int bh = wid >> 5;
  const int b = bh >> 4, h = bh & 15;
  const int s0 = chunk * 128;

  const float* Kb = K + (size_t)b * S_ * DM_ + h * D_;
  const float* Vb = V + (size_t)b * S_ * DM_ + h * D_;

  v8f acc[4][4] = {};
  for (int s = s0; s < s0 + 128; s += 4) {
    const int ks = s + lh * 2;     // lane<16 -> K=s,s+1 ; lane>=16 -> s+2,s+3
    v2f a[4], bb[4];
#pragma unroll
    for (int i = 0; i < 4; ++i) {  // A[m=d][k=s] = K[s][d]
      const float* p = Kb + (size_t)ks * DM_ + i * 16 + l15;
      v2f t = { p[0], p[DM_] }; a[i] = t;
    }
#pragma unroll
    for (int j = 0; j < 4; ++j) {  // B[k=s][n=e] = V[s][e]
      const float* p = Vb + (size_t)ks * DM_ + j * 16 + l15;
      v2f t = { p[0], p[DM_] }; bb[j] = t;
    }
#pragma unroll
    for (int i = 0; i < 4; ++i)
#pragma unroll
      for (int j = 0; j < 4; ++j)
        acc[i][j] = __builtin_amdgcn_wmma_f32_16x16x4_f32(
            false, a[i], false, bb[j], (short)0, acc[i][j], false, false);
  }
  float* kvb = kv + (size_t)bh * (D_ * D_);
#pragma unroll
  for (int i = 0; i < 4; ++i)
#pragma unroll
    for (int j = 0; j < 4; ++j)
#pragma unroll
      for (int r = 0; r < 8; ++r)
        atomicAdd(&kvb[(i * 16 + r + 8 * lh) * D_ + j * 16 + l15], acc[i][j][r]);
}

// ---------------- per-row normalizer: q . ksum + eps + msum ----------------
__global__ void __launch_bounds__(256) k_qnorm(
    const float* __restrict__ Q, const float* __restrict__ ksum,
    const float* __restrict__ msum, float* __restrict__ nrm)
{
  const int lane = threadIdx.x & 31;
  const int warp = (blockIdx.x * 256 + threadIdx.x) >> 5;
  const int h  = warp & (H_ - 1);
  const int bs = warp >> 4;
  const int b  = bs >> 12;
  const int s  = bs & (S_ - 1);
  const int bh = (b << 4) | h;

  const float2 q  = *(const float2*)(Q + (size_t)bs * DM_ + h * D_ + lane * 2);
  const float2 ks = *(const float2*)(ksum + bh * D_ + lane * 2);
  float d = q.x * ks.x + q.y * ks.y;
#pragma unroll
  for (int o = 16; o > 0; o >>= 1) d += __shfl_xor(d, o, 32);
  if (lane == 0) nrm[(size_t)bh * S_ + s] = d + EPS_ + msum[b];
}

// ---------------- ctx = Q @ KV + vsum, divided by nrm ----------------
// wave = (bh, s-tile of 16): 16x64 output tile via f32 WMMA over D=64.
__global__ void __launch_bounds__(256) k_ctx(
    const float* __restrict__ Q, const float* __restrict__ kv,
    const float* __restrict__ vsum, const float* __restrict__ nrm,
    float* __restrict__ out)
{
  const int lane = threadIdx.x & 31;
  const int l15 = lane & 15, lh = lane >> 4;
  const int wid = blockIdx.x * 8 + (threadIdx.x >> 5);  // < 64*256
  const int st = wid & 255;
  const int bh = wid >> 8;
  const int b = bh >> 4, h = bh & 15;
  const int s0 = st * 16;

  const float* Qb  = Q  + ((size_t)b * S_ + s0) * DM_ + h * D_;
  const float* kvb = kv + (size_t)bh * (D_ * D_);

  v8f acc[4] = {};
  for (int d0 = 0; d0 < D_; d0 += 4) {
    const float* pa = Qb + (size_t)l15 * DM_ + d0 + lh * 2;  // A[m=s][k=d]
    v2f a = { pa[0], pa[1] };
#pragma unroll
    for (int j = 0; j < 4; ++j) {                            // B[k=d][n=e]
      const float* pb = kvb + (d0 + lh * 2) * D_ + j * 16 + l15;
      v2f bb = { pb[0], pb[D_] };
      acc[j] = __builtin_amdgcn_wmma_f32_16x16x4_f32(
          false, a, false, bb, (short)0, acc[j], false, false);
    }
  }

  float* ob = out + ((size_t)b * S_ + s0) * DM_ + h * D_;
  const float* vs = vsum + bh * D_;
  const float* nb = nrm + (size_t)bh * S_ + s0;
#pragma unroll
  for (int j = 0; j < 4; ++j) {
    const int e = j * 16 + l15;
    const float vse = vs[e];
#pragma unroll
    for (int r = 0; r < 8; ++r) {
      const int m = r + 8 * lh;
      ob[(size_t)m * DM_ + e] = (acc[j][r] + vse) / nb[m];
    }
  }
}

// ---------------- host launcher ----------------
extern "C" void kernel_launch(void* const* d_in, const int* in_sizes, int n_in,
                              void* d_out, int out_size, void* d_ws, size_t ws_size,
                              hipStream_t stream) {
  const float* X  = (const float*)d_in[0];
  const float* am = (const float*)d_in[1];
  const float* Wq = (const float*)d_in[2];
  const float* bq = (const float*)d_in[3];
  const float* Wk = (const float*)d_in[4];
  const float* bk = (const float*)d_in[5];
  const float* Wv = (const float*)d_in[6];
  const float* bv = (const float*)d_in[7];
  float* out = (float*)d_out;

  const size_t MiB = 1u << 20;
  char* ws = (char*)d_ws;
  bf16_t* Xbf = (bf16_t*)(ws);                     // 16384*1024 bf16 = 32 MiB
  bf16_t* Wt  = (bf16_t*)(ws + 32 * MiB);          // 3 * 2 MiB
  float*  Q   = (float*)(ws + 38 * MiB);           // 64 MiB
  float*  K   = (float*)(ws + 102 * MiB);          // 64 MiB
  float*  V   = (float*)(ws + 166 * MiB);          // 64 MiB
  float*  kv  = (float*)(ws + 230 * MiB);          // 64*64*64 f32 = 1 MiB
  float*  ksum = kv + 64 * D_ * D_;                // 64*64
  float*  vsum = ksum + 64 * D_;                   // 64*64
  float*  msum = vsum + 64 * D_;                   // 4
  float*  nrm  = (float*)(ws + 232 * MiB);         // 64*4096 f32 = 1 MiB

  const int nX = B_ * S_ * DM_;                    // 16,777,216

  // zero the accumulator region (kv, ksum, vsum, msum are contiguous)
  const int nz = 64 * D_ * D_ + 2 * 64 * D_ + B_;
  k_zero<<<(nz + 255) / 256, 256, 0, stream>>>(kv, nz);

  // precision conversion
  k_cvt<<<nX / 256, 256, 0, stream>>>(X, Xbf, nX);
  k_cvt_t<<<4096, 256, 0, stream>>>(Wq, Wt);
  k_cvt_t<<<4096, 256, 0, stream>>>(Wk, Wt + (size_t)DM_ * DM_);
  k_cvt_t<<<4096, 256, 0, stream>>>(Wv, Wt + 2 * (size_t)DM_ * DM_);

  // fused QKV projection: 3*16*256 = 12288 wave-tiles, 8 waves/block
  k_gemm_qkv<<<1536, 256, 0, stream>>>(Xbf, Wt, bq, bk, bv, Q, K, V);

  // mask + cosine normalization + row-sum accumulators (262144 waves)
  k_normalize<<<32768, 256, 0, stream>>>(Q, K, V, am, ksum, vsum, msum);

  // KV = K^T V per head (64 bh x 32 chunks = 2048 waves)
  k_kv_accum<<<256, 256, 0, stream>>>(K, V, kv);

  // per-row normalizer
  k_qnorm<<<32768, 256, 0, stream>>>(Q, ksum, msum, nrm);

  // ctx + output (64 bh x 256 s-tiles = 16384 waves)
  k_ctx<<<2048, 256, 0, stream>>>(Q, kv, vsum, nrm, out);
}
